// StagePolicyNetwork_12979391169442
// MI455X (gfx1250) — matrix operand order repaired and checked by reading, not compiled, not run on hardware
//
#include <hip/hip_runtime.h>
#include <hip/hip_bf16.h>

typedef float  v8f   __attribute__((ext_vector_type(8)));
typedef __bf16 v16bf __attribute__((ext_vector_type(16)));
typedef __bf16 v8bf  __attribute__((ext_vector_type(8)));

#define KIN     416
#define HDIM    1024
#define ROWS    64
#define LDA     424      // input tile row stride (elems); 848B, 16B-aligned, conflict-free
#define LDH     1032     // h1 row stride (elems); 2064B, 16B-aligned, breaks bank alignment
#define THREADS 512      // 16 waves (wave32)

// ---------------- weight prep: transpose + f32->bf16 into d_ws ----------------
__global__ void prep_weights(const float* __restrict__ W1, const float* __restrict__ W2,
                             __bf16* __restrict__ w1t, __bf16* __restrict__ w2t) {
  int idx = blockIdx.x * blockDim.x + threadIdx.x;
  const int n1 = KIN * HDIM;
  if (idx < n1) {
    int k = idx / HDIM, n = idx - k * HDIM;          // coalesced read of W1[k][n]
    w1t[(size_t)n * KIN + k] = (__bf16)W1[idx];
  } else {
    idx -= n1;
    if (idx < HDIM * HDIM) {
      int k = idx / HDIM, n = idx - k * HDIM;
      w2t[(size_t)n * HDIM + k] = (__bf16)W2[idx];
    }
  }
}

// ---------------- fused gather + 3-layer MLP ----------------
__global__ __launch_bounds__(THREADS, 1)
void mlp_fused(const float* __restrict__ x, const float* __restrict__ h_node,
               const float* __restrict__ h_dag, const float* __restrict__ h_glob,
               const long long* __restrict__ batch, const long long* __restrict__ stage_idx,
               const __bf16* __restrict__ w1t, const float* __restrict__ b1,
               const __bf16* __restrict__ w2t, const float* __restrict__ b2,
               const float* __restrict__ W3, const float* __restrict__ b3,
               float* __restrict__ out)
{
  extern __shared__ char smem_raw[];
  __bf16* A0     = (__bf16*)smem_raw;                                     // [ROWS][LDA] bf16
  __bf16* H1     = (__bf16*)(smem_raw + (size_t)ROWS * LDA * 2);          // [ROWS][LDH] bf16
  float*  scores = (float*)(smem_raw + (size_t)ROWS * LDA * 2 + (size_t)ROWS * LDH * 2);

  __shared__ long long rowStage[ROWS];
  __shared__ int       rowG[ROWS];

  const int tid   = threadIdx.x;
  const int mBase = blockIdx.x * ROWS;

  if (tid < ROWS) {
    long long s   = stage_idx[mBase + tid];
    rowStage[tid] = s;
    rowG[tid]     = (int)batch[s];
    scores[tid]   = 0.0f;
  }
  __syncthreads();

  // Build gathered/concatenated input tile [ROWS x 416] as bf16 in LDS.
  for (int idx = tid; idx < ROWS * KIN; idx += THREADS) {
    int r = idx / KIN;
    int c = idx - r * KIN;
    long long s = rowStage[r];
    int g = rowG[r];
    float v;
    if (c < 32)        v = x[s * 32 + c];
    else if (c < 160)  v = h_node[s * 128 + (c - 32)];
    else if (c < 288)  v = h_dag[(size_t)g * 128 + (c - 160)];
    else               v = h_glob[(size_t)g * 128 + (c - 288)];
    A0[r * LDA + c] = (__bf16)v;
  }
  __syncthreads();

  const int wave  = tid >> 5;
  const int lane  = tid & 31;
  const int l16   = lane & 15;
  const int khalf = lane >> 4;
  const int nb    = wave * 64;          // this wave's 64-column slice of HDIM

  // ---------------- layer 1: H1 = relu(A0 @ W1 + b1) ----------------
  {
    v8f c[4][4] = {};
    for (int ks = 0; ks < KIN / 32; ++ks) {
      union { v16bf v; v8bf h[2]; } a[4];
      #pragma unroll
      for (int i = 0; i < 4; ++i) {
        const __bf16* p = &A0[(i * 16 + l16) * LDA + ks * 32 + khalf * 8];
        a[i].h[0] = *(const v8bf*)p;
        a[i].h[1] = *(const v8bf*)(p + 16);
      }
      #pragma unroll
      for (int j = 0; j < 4; ++j) {
        const __bf16* q = &w1t[(size_t)(nb + j * 16 + l16) * KIN + ks * 32 + khalf * 8];
        union { v16bf v; v8bf h[2]; } bu;
        bu.h[0] = *(const v8bf*)q;
        bu.h[1] = *(const v8bf*)(q + 16);
        v16bf b = bu.v;
        #pragma unroll
        for (int i = 0; i < 4; ++i)
          c[i][j] = __builtin_amdgcn_wmma_f32_16x16x32_bf16(
              false, a[i].v, false, b, (short)0, c[i][j], false, false);
      }
    }
    // epilogue: bias + relu + bf16 -> LDS
    #pragma unroll
    for (int j = 0; j < 4; ++j) {
      int n = nb + j * 16 + l16;
      float bias = b1[n];
      #pragma unroll
      for (int i = 0; i < 4; ++i) {
        #pragma unroll
        for (int r = 0; r < 8; ++r) {
          int m = i * 16 + r + khalf * 8;
          float v = c[i][j][r] + bias;
          H1[m * LDH + n] = (__bf16)(v > 0.0f ? v : 0.0f);
        }
      }
    }
  }
  __syncthreads();

  // ---------------- layer 2 + layer 3 fused ----------------
  {
    v8f c[4][4] = {};
    for (int ks = 0; ks < HDIM / 32; ++ks) {
      union { v16bf v; v8bf h[2]; } a[4];
      #pragma unroll
      for (int i = 0; i < 4; ++i) {
        const __bf16* p = &H1[(i * 16 + l16) * LDH + ks * 32 + khalf * 8];
        a[i].h[0] = *(const v8bf*)p;
        a[i].h[1] = *(const v8bf*)(p + 16);
      }
      #pragma unroll
      for (int j = 0; j < 4; ++j) {
        const __bf16* q = &w2t[(size_t)(nb + j * 16 + l16) * HDIM + ks * 32 + khalf * 8];
        union { v16bf v; v8bf h[2]; } bu;
        bu.h[0] = *(const v8bf*)q;
        bu.h[1] = *(const v8bf*)(q + 16);
        v16bf b = bu.v;
        #pragma unroll
        for (int i = 0; i < 4; ++i)
          c[i][j] = __builtin_amdgcn_wmma_f32_16x16x32_bf16(
              false, a[i].v, false, b, (short)0, c[i][j], false, false);
      }
    }
    // epilogue: score[m] += sum_n relu(h2[m][n] + b2[n]) * W3[n]
    float part[4][8];
    #pragma unroll
    for (int i = 0; i < 4; ++i)
      #pragma unroll
      for (int r = 0; r < 8; ++r)
        part[i][r] = 0.0f;

    #pragma unroll
    for (int j = 0; j < 4; ++j) {
      int n = nb + j * 16 + l16;
      float bias = b2[n];
      float wv   = W3[n];
      #pragma unroll
      for (int i = 0; i < 4; ++i)
        #pragma unroll
        for (int r = 0; r < 8; ++r) {
          float v = c[i][j][r] + bias;
          part[i][r] += (v > 0.0f ? v : 0.0f) * wv;
        }
    }
    #pragma unroll
    for (int i = 0; i < 4; ++i)
      #pragma unroll
      for (int r = 0; r < 8; ++r)
        atomicAdd(&scores[i * 16 + r + khalf * 8], part[i][r]);
  }
  __syncthreads();

  if (tid < ROWS) out[mBase + tid] = scores[tid] + b3[0];
}

extern "C" void kernel_launch(void* const* d_in, const int* in_sizes, int n_in,
                              void* d_out, int out_size, void* d_ws, size_t ws_size,
                              hipStream_t stream) {
  const float*     x         = (const float*)d_in[0];
  const float*     h_node    = (const float*)d_in[1];
  const float*     h_dag     = (const float*)d_in[2];
  const float*     h_glob    = (const float*)d_in[3];
  const long long* batch     = (const long long*)d_in[4];
  const long long* stage_idx = (const long long*)d_in[5];
  // d_in[6] = num_stage_acts (consistent with batch[stage_idx]; unused)
  const float*     W1 = (const float*)d_in[7];
  const float*     b1 = (const float*)d_in[8];
  const float*     W2 = (const float*)d_in[9];
  const float*     b2 = (const float*)d_in[10];
  const float*     W3 = (const float*)d_in[11];
  const float*     b3 = (const float*)d_in[12];
  float*           out = (float*)d_out;

  __bf16* w1t = (__bf16*)d_ws;
  __bf16* w2t = (__bf16*)((char*)d_ws + (size_t)KIN * HDIM * 2);

  const int prepN = KIN * HDIM + HDIM * HDIM;
  prep_weights<<<(prepN + 255) / 256, 256, 0, stream>>>(W1, W2, w1t, w2t);

  const int M = in_sizes[5];                    // 131072 stage rows
  const size_t smem = (size_t)ROWS * LDA * 2 + (size_t)ROWS * LDH * 2 + (size_t)ROWS * 4;
  mlp_fused<<<M / ROWS, THREADS, smem, stream>>>(x, h_node, h_dag, h_glob, batch, stage_idx,
                                                 w1t, b1, w2t, b2, W3, b3, out);
}
